// HunYuanMoEV1Moe_41850161332333
// MI455X (gfx1250) — compile-verified, matrix operand order checked by reading, not run on hardware
//
#include <hip/hip_runtime.h>
#include <hip/hip_bf16.h>

typedef __bf16 bf16;
typedef __attribute__((ext_vector_type(16))) __bf16 v16bf;
typedef __attribute__((ext_vector_type(8)))  __bf16 v8bf;
typedef __attribute__((ext_vector_type(2)))  __bf16 v2bf;
typedef __attribute__((ext_vector_type(8)))  float  v8f;
typedef __attribute__((ext_vector_type(4)))  float  v4f;

#define T_TOK 4096   // B*S tokens
#define HID   2048   // hidden size
#define INTER 4096   // intermediate size
#define NEXP  8      // experts

// GEMM tiling: 256 threads = 8 waves (wave32). Wave grid 4x2, each wave owns
// 32x32 of the 128x64 output tile => 2x2 WMMA 16x16 accumulators per matrix.
#define BM 128
#define BN 64
#define BK 32
#define LDP 40   // LDS row stride (bf16): 80B, 16B-aligned, conflict-breaking pad

__device__ inline v8f zero8() {
  v8f z;
#pragma unroll
  for (int i = 0; i < 8; i++) z[i] = 0.f;
  return z;
}

// ---- CDNA5 async global->LDS DMA (ASYNCcnt) -------------------------------
// global_load_async_to_lds_b128: VDST = VGPR holding LDS byte address,
// VADDR = per-lane 32-bit byte offset, SADDR = uniform 64-bit base (GVS mode).
__device__ inline void async_ld_b128(unsigned lds_addr, const void* sbase,
                                     unsigned byte_off) {
  asm volatile("global_load_async_to_lds_b128 %0, %1, %2"
               :: "v"(lds_addr), "v"(byte_off), "s"(sbase)
               : "memory");
}
__device__ inline void wait_async0() {
  asm volatile("s_wait_asynccnt 0x0" ::: "memory");
}
// generic LDS pointer -> 32-bit LDS byte address (low 32 bits of flat addr)
__device__ inline unsigned lds_addr_of(const void* p) {
  return (unsigned)(uintptr_t)p;
}

// Build a 16x32 bf16 WMMA operand fragment from an LDS tile (row-major,
// row stride LDP). ISA layout: lane L -> row = L%16, half = L/16;
// v16 elems [0..7] = K = half*8.., [8..15] = K = 16+half*8.. ; both chunks
// are 16B-contiguous => two ds_load_b128 per fragment.
__device__ inline v16bf frag_ld(const bf16* p, int lane) {
  int r = lane & 15, h2 = lane >> 4;
  const v8bf* a = (const v8bf*)(p + r * LDP + h2 * 8);
  const v8bf* b = (const v8bf*)(p + r * LDP + 16 + h2 * 8);
  v8bf lo = *a, hi = *b;
  v16bf f;
#pragma unroll
  for (int i = 0; i < 8; i++) { f[i] = lo[i]; f[i + 8] = hi[i]; }
  return f;
}

__device__ inline v8f wmma_bf16(v16bf a, v16bf b, v8f c) {
  return __builtin_amdgcn_wmma_f32_16x16x32_bf16(false, a, false, b, (short)0, c,
                                                 false, false);
}

// ---------------------------------------------------------------- router ----
__global__ __launch_bounds__(256)
void router_kernel(const float* __restrict__ x, const float* __restrict__ wg,
                   int* __restrict__ sel, float* __restrict__ selw) {
  int lane = threadIdx.x & 31, wave = threadIdx.x >> 5;
  int t = blockIdx.x * 8 + wave;
  const float* xr = x + (size_t)t * HID;
  float acc[NEXP];
#pragma unroll
  for (int e = 0; e < NEXP; e++) acc[e] = 0.f;
  for (int k = lane; k < HID; k += 32) {
    float xv = xr[k];
#pragma unroll
    for (int e = 0; e < NEXP; e++) acc[e] += xv * wg[e * HID + k];
  }
#pragma unroll
  for (int e = 0; e < NEXP; e++)
#pragma unroll
    for (int off = 16; off > 0; off >>= 1) acc[e] += __shfl_xor(acc[e], off, 32);

  if (lane == 0) {
    int a = 0;
#pragma unroll
    for (int e = 1; e < NEXP; e++) if (acc[e] > acc[a]) a = e;  // first-max tie-break
    int b = (a == 0) ? 1 : 0;
#pragma unroll
    for (int e = 0; e < NEXP; e++) if (e != a && acc[e] > acc[b]) b = e;
    float wa = 1.f / (1.f + __expf(acc[b] - acc[a]));
    sel[t * 2] = a;  sel[t * 2 + 1] = b;
    selw[t * 2] = wa; selw[t * 2 + 1] = 1.f - wa;
  }
}

// ---------------------------------------------------- deterministic lists ----
__global__ __launch_bounds__(256)
void build_lists(const int* __restrict__ sel, const float* __restrict__ selw,
                 int* __restrict__ idxl, float* __restrict__ wll,
                 int* __restrict__ cnt) {
  int e = blockIdx.x;
  int tid = threadIdx.x, lane = tid & 31, wave = tid >> 5;
  __shared__ int wsum[8];
  int base = 0;
  for (int t0 = 0; t0 < T_TOK; t0 += 256) {
    int t = t0 + tid;
    int s0 = sel[t * 2], s1 = sel[t * 2 + 1];
    bool f0 = (s0 == e);
    bool f = f0 || (s1 == e);
    float w = f0 ? selw[t * 2] : selw[t * 2 + 1];
    unsigned mask = (unsigned)__ballot(f);
    int prefix = __popc(mask & ((1u << lane) - 1u));
    if (lane == 0) wsum[wave] = __popc(mask);
    __syncthreads();
    int woff = 0, tot = 0;
#pragma unroll
    for (int i = 0; i < 8; i++) { int s = wsum[i]; tot += s; if (i < wave) woff += s; }
    if (f) {
      int pos = base + woff + prefix;
      idxl[(size_t)e * T_TOK + pos] = t;
      wll[(size_t)e * T_TOK + pos] = w;
    }
    base += tot;
    __syncthreads();
  }
  if (tid == 0) cnt[e] = base;
}

// ------------------------------------------------------------- fp32->bf16 ----
__global__ __launch_bounds__(256)
void cvt_kernel(const float* __restrict__ x, bf16* __restrict__ xb) {
  int i = (blockIdx.x * 256 + threadIdx.x) * 4;
  v4f v = *(const v4f*)(x + i);
#pragma unroll
  for (int j = 0; j < 4; j++) xb[i + j] = (bf16)v[j];
}

// ------------------------------------------------------------ gate+up GEMM ---
// Double-buffered: tile k+1 is prefetched (A via async-to-LDS DMA, B via
// fp32 load + bf16 convert + packed v2bf stores) while WMMAs consume tile k.
__global__ __launch_bounds__(256)
void gateup_kernel(const bf16* __restrict__ xb, const float* __restrict__ gw,
                   const float* __restrict__ uw, bf16* __restrict__ hbuf,
                   const int* __restrict__ idxl, const int* __restrict__ cnt) {
  __shared__ bf16 lsA[2][BM * LDP];
  __shared__ bf16 lsG[2][BN * LDP];
  __shared__ bf16 lsU[2][BN * LDP];
  __shared__ int lsTok[BM];

  int M = cnt ? *cnt : T_TOK;
  int m0 = blockIdx.x * BM;
  if (m0 >= M) return;                    // early-exit unused capacity tiles
  int n0 = blockIdx.y * BN;
  int tid = threadIdx.x;

  if (tid < BM) {
    int m = m0 + tid;
    lsTok[tid] = idxl ? ((m < M) ? idxl[m] : 0) : m;
  }
  __syncthreads();

  int lane = tid & 31, wave = tid >> 5;
  int wm = wave & 3, wn = wave >> 2;

  // A tile: 128x32 bf16 gathered rows, 2x b128 async DMA per thread
  auto loadA = [&](int buf, int k0) {
#pragma unroll
    for (int j = 0; j < 2; j++) {
      int u = tid + 256 * j;
      int r = u >> 2, c4 = u & 3;
      unsigned goff = (unsigned)(((unsigned)lsTok[r] * HID + k0 + c4 * 8) * 2);
      async_ld_b128(lds_addr_of(&lsA[buf][r * LDP + c4 * 8]), xb, goff);
    }
  };
  // B tiles: each thread owns a (k,k+1)x4n micro-tile of both gate and up,
  // emitting packed 2xbf16 stores into the transposed [n][k] LDS layout.
  auto loadB = [&](int buf, int k0) {
    int nf = tid & 15, kp = tid >> 4;
    int k = kp * 2;
    size_t go = (size_t)(k0 + k) * INTER + n0 + nf * 4;
    v4f g0 = *(const v4f*)(gw + go);
    v4f g1 = *(const v4f*)(gw + go + INTER);
    v4f u0 = *(const v4f*)(uw + go);
    v4f u1 = *(const v4f*)(uw + go + INTER);
#pragma unroll
    for (int i = 0; i < 4; i++) {
      v2bf pg = {(bf16)g0[i], (bf16)g1[i]};
      v2bf pu = {(bf16)u0[i], (bf16)u1[i]};
      *(v2bf*)&lsG[buf][(nf * 4 + i) * LDP + k] = pg;
      *(v2bf*)&lsU[buf][(nf * 4 + i) * LDP + k] = pu;
    }
  };

  v8f accG[2][2], accU[2][2];
#pragma unroll
  for (int i = 0; i < 2; i++)
#pragma unroll
    for (int j = 0; j < 2; j++) { accG[i][j] = zero8(); accU[i][j] = zero8(); }

  int buf = 0;
  loadA(0, 0);
  loadB(0, 0);
  wait_async0();
  __syncthreads();

  for (int k0 = 0; k0 < HID; k0 += BK) {
    bool more = (k0 + BK) < HID;
    if (more) { loadA(buf ^ 1, k0 + BK); loadB(buf ^ 1, k0 + BK); }

    v16bf af[2], bg[2], bu[2];
#pragma unroll
    for (int i = 0; i < 2; i++)
      af[i] = frag_ld(&lsA[buf][(wm * 32 + i * 16) * LDP], lane);
#pragma unroll
    for (int i = 0; i < 2; i++) {
      bg[i] = frag_ld(&lsG[buf][(wn * 32 + i * 16) * LDP], lane);
      bu[i] = frag_ld(&lsU[buf][(wn * 32 + i * 16) * LDP], lane);
    }
#pragma unroll
    for (int i = 0; i < 2; i++)
#pragma unroll
      for (int jn = 0; jn < 2; jn++) {
        accG[i][jn] = wmma_bf16(af[i], bg[jn], accG[i][jn]);
        accU[i][jn] = wmma_bf16(af[i], bu[jn], accU[i][jn]);
      }
    if (more) {
      wait_async0();
      __syncthreads();
      buf ^= 1;
    }
  }

  // epilogue: h = silu(g)*u with fast v_rcp_f32 sigmoid (output is bf16)
  int hl = lane >> 4, col = lane & 15;
#pragma unroll
  for (int i = 0; i < 2; i++)
#pragma unroll
    for (int jn = 0; jn < 2; jn++)
#pragma unroll
      for (int r = 0; r < 8; r++) {
        float g = accG[i][jn][r];
        float u = accU[i][jn][r];
        float hv = g * __builtin_amdgcn_rcpf(1.f + __expf(-g)) * u;
        int ml = wm * 32 + i * 16 + hl * 8 + r;
        int n = n0 + wn * 32 + jn * 16 + col;
        hbuf[(size_t)(m0 + ml) * INTER + n] = (bf16)hv;
      }
}

// ---------------------------------------------------------------- down GEMM --
__global__ __launch_bounds__(256)
void down_kernel(const bf16* __restrict__ hbuf, const float* __restrict__ dw,
                 float* __restrict__ out, const int* __restrict__ idxl,
                 const float* __restrict__ wl, const int* __restrict__ cnt) {
  __shared__ bf16 lsA[2][BM * LDP];
  __shared__ bf16 lsB[2][BN * LDP];
  __shared__ int lsTok[BM];
  __shared__ float lsW[BM];

  int M = cnt ? *cnt : T_TOK;
  int m0 = blockIdx.x * BM;
  if (m0 >= M) return;
  int n0 = blockIdx.y * BN;
  int tid = threadIdx.x;
  if (tid < BM) {
    int m = m0 + tid;
    bool v = m < M;
    lsTok[tid] = idxl ? (v ? idxl[m] : 0) : m;
    lsW[tid]   = wl   ? (v ? wl[m] : 0.f) : 1.f;
  }
  __syncthreads();

  int lane = tid & 31, wave = tid >> 5;
  int wm = wave & 3, wn = wave >> 2;

  auto loadA = [&](int buf, int k0) {
#pragma unroll
    for (int j = 0; j < 2; j++) {
      int u = tid + 256 * j;
      int r = u >> 2, c4 = u & 3;
      unsigned goff = (unsigned)((((unsigned)(m0 + r)) * INTER + k0 + c4 * 8) * 2);
      async_ld_b128(lds_addr_of(&lsA[buf][r * LDP + c4 * 8]), hbuf, goff);
    }
  };
  auto loadB = [&](int buf, int k0) {
    int nf = tid & 15, kp = tid >> 4;
    int k = kp * 2;
    size_t go = (size_t)(k0 + k) * HID + n0 + nf * 4;
    v4f b0 = *(const v4f*)(dw + go);
    v4f b1 = *(const v4f*)(dw + go + HID);
#pragma unroll
    for (int i = 0; i < 4; i++) {
      v2bf pb = {(bf16)b0[i], (bf16)b1[i]};
      *(v2bf*)&lsB[buf][(nf * 4 + i) * LDP + k] = pb;
    }
  };

  v8f acc[2][2];
#pragma unroll
  for (int i = 0; i < 2; i++)
#pragma unroll
    for (int j = 0; j < 2; j++) acc[i][j] = zero8();

  int buf = 0;
  loadA(0, 0);
  loadB(0, 0);
  wait_async0();
  __syncthreads();

  for (int k0 = 0; k0 < INTER; k0 += BK) {
    bool more = (k0 + BK) < INTER;
    if (more) { loadA(buf ^ 1, k0 + BK); loadB(buf ^ 1, k0 + BK); }

    v16bf af[2], bb[2];
#pragma unroll
    for (int i = 0; i < 2; i++)
      af[i] = frag_ld(&lsA[buf][(wm * 32 + i * 16) * LDP], lane);
#pragma unroll
    for (int i = 0; i < 2; i++)
      bb[i] = frag_ld(&lsB[buf][(wn * 32 + i * 16) * LDP], lane);
#pragma unroll
    for (int i = 0; i < 2; i++)
#pragma unroll
      for (int jn = 0; jn < 2; jn++) acc[i][jn] = wmma_bf16(af[i], bb[jn], acc[i][jn]);

    if (more) {
      wait_async0();
      __syncthreads();
      buf ^= 1;
    }
  }

  int hl = lane >> 4, col = lane & 15;
#pragma unroll
  for (int i = 0; i < 2; i++)
#pragma unroll
    for (int jn = 0; jn < 2; jn++)
#pragma unroll
      for (int r = 0; r < 8; r++) {
        int ml = wm * 32 + i * 16 + hl * 8 + r;
        int m = m0 + ml;
        if (m < M) {
          int n = n0 + wn * 32 + jn * 16 + col;
          float* op = out + (size_t)lsTok[ml] * HID + n;
          float v = acc[i][jn][r];
          if (idxl) *op += lsW[ml] * v;   // expert pass: accumulate (stream-ordered)
          else      *op = v;              // shared pass: first full write of out
        }
      }
}

// ------------------------------------------------------------------ launch ---
extern "C" void kernel_launch(void* const* d_in, const int* in_sizes, int n_in,
                              void* d_out, int out_size, void* d_ws, size_t ws_size,
                              hipStream_t stream) {
  (void)in_sizes; (void)n_in; (void)out_size; (void)ws_size;
  const float* x   = (const float*)d_in[0];
  const float* wg  = (const float*)d_in[1];
  const float* gw  = (const float*)d_in[2];
  const float* uw  = (const float*)d_in[3];
  const float* dwn = (const float*)d_in[4];
  const float* sgw = (const float*)d_in[5];
  const float* suw = (const float*)d_in[6];
  const float* sdw = (const float*)d_in[7];
  float* out = (float*)d_out;

  char* ws = (char*)d_ws;
  bf16* xb   = (bf16*)ws;                                    // 16 MiB
  bf16* hbuf = (bf16*)(ws + (size_t)T_TOK * HID * 2);        // 32 MiB
  char* p = ws + (size_t)T_TOK * HID * 2 + (size_t)T_TOK * INTER * 2;
  int*   sel  = (int*)p;   p += (size_t)T_TOK * 2 * sizeof(int);
  float* selw = (float*)p; p += (size_t)T_TOK * 2 * sizeof(float);
  int*   idxl = (int*)p;   p += (size_t)NEXP * T_TOK * sizeof(int);
  float* wll  = (float*)p; p += (size_t)NEXP * T_TOK * sizeof(float);
  int*   cnt  = (int*)p;

  cvt_kernel<<<(T_TOK * HID) / (4 * 256), 256, 0, stream>>>(x, xb);
  router_kernel<<<T_TOK / 8, 256, 0, stream>>>(x, wg, sel, selw);
  build_lists<<<NEXP, 256, 0, stream>>>(sel, selw, idxl, wll, cnt);

  dim3 gu(T_TOK / BM, INTER / BN);
  dim3 gd(T_TOK / BM, HID / BN);

  // shared expert: all tokens, writes out
  gateup_kernel<<<gu, 256, 0, stream>>>(xb, sgw, suw, hbuf, nullptr, nullptr);
  down_kernel<<<gd, 256, 0, stream>>>(hbuf, sdw, out, nullptr, nullptr, nullptr);

  // routed experts: compacted tokens, accumulate into out
  for (int e = 0; e < NEXP; e++) {
    gateup_kernel<<<gu, 256, 0, stream>>>(
        xb, gw + (size_t)e * HID * INTER, uw + (size_t)e * HID * INTER, hbuf,
        idxl + (size_t)e * T_TOK, cnt + e);
    down_kernel<<<gd, 256, 0, stream>>>(
        hbuf, dwn + (size_t)e * INTER * HID, out, idxl + (size_t)e * T_TOK,
        wll + (size_t)e * T_TOK, cnt + e);
  }
}
